// TreeAttentionV2_62543313764992
// MI455X (gfx1250) — matrix operand order repaired and checked by reading, not compiled, not run on hardware
//
#include <hip/hip_runtime.h>

// ---------------------------------------------------------------------------
// Fused attention block for MI455X (gfx1250, wave32, WMMA).
// fp32 inputs -> bf16 once; all GEMM stages on v_wmma_f32_16x16x32_bf16
// (fp32 accumulate). Tile staging uses CDNA5 async global->LDS DMA
// (global_load_async_to_lds_b128, ASYNCcnt) with double-buffered LDS and
// one-deep prefetch so DMA latency overlaps the WMMA stream. Post-softmax
// threshold forces a two-pass softmax (Z first, then P = relu(exp*Z^-1 - t)).
// ---------------------------------------------------------------------------

typedef unsigned short u16;
typedef __attribute__((ext_vector_type(16))) __bf16 v16bf;
typedef __attribute__((ext_vector_type(8)))  float  v8f;

#define B_  4
#define T_  2048
#define C_  1024
#define H_  16
#define D_  64
#define SCALE 0.125f        // 1/sqrt(64)
#define THRESH 0.001f

// workspace layout (u16 elements)
constexpr size_t N_X   = (size_t)B_ * T_ * C_;        // 8388608
constexpr size_t N_WA  = (size_t)C_ * 3 * C_;         // 3145728
constexpr size_t N_WP  = (size_t)C_ * C_;             // 1048576
constexpr size_t N_HED = (size_t)B_ * H_ * T_ * D_;   // 8388608
constexpr size_t OFF_XB  = 0;
constexpr size_t OFF_WAT = OFF_XB  + N_X;
constexpr size_t OFF_WPT = OFF_WAT + N_WA;
constexpr size_t OFF_Q   = OFF_WPT + N_WP;
constexpr size_t OFF_K   = OFF_Q   + N_HED;
constexpr size_t OFF_VT  = OFF_K   + N_HED;
constexpr size_t OFF_AO  = OFF_VT  + N_HED;           // total ~88 MB

union Frag { uint4 q[2]; v16bf v; };                  // 32B A/B operand

__device__ __forceinline__ u16 f2bf(float f) {
  unsigned int u = __builtin_bit_cast(unsigned int, f);
  u += 0x7FFFu + ((u >> 16) & 1u);                    // round-to-nearest-even
  return (u16)(u >> 16);
}

__device__ __forceinline__ v8f wmma_bf16(const Frag& a, const Frag& b, v8f c) {
  return __builtin_amdgcn_wmma_f32_16x16x32_bf16(false, a.v, false, b.v,
                                                 (short)0, c, false, false);
}

// CDNA5 async DMA: 16B global -> LDS, tracked by ASYNCcnt (ISA 15.18.3 op 98).
__device__ __forceinline__ void async_b128(unsigned lds_off, const void* g) {
  asm volatile("global_load_async_to_lds_b128 %0, %1, off"
               :: "v"(lds_off), "v"(g) : "memory");
}
__device__ __forceinline__ void wait_async0() {
  asm volatile("s_wait_asynccnt 0" ::: "memory");
}
__device__ __forceinline__ unsigned lds_off(const void* p) {
  return (unsigned)(size_t)p;                          // LDS aperture: low 32 bits
}

// ---------------------------------------------------------------- converts --
__global__ __launch_bounds__(256) void cvt_f32_bf16(const float* __restrict__ s,
                                                    u16* __restrict__ d, size_t n) {
  size_t i = (size_t)blockIdx.x * 256 + threadIdx.x;
  if (i < n) d[i] = f2bf(s[i]);
}

// src [C_ x nOut] row-major -> dst [nOut x C_] bf16 (B^T so K is contiguous)
__global__ __launch_bounds__(256) void cvt_transpose(const float* __restrict__ s,
                                                     u16* __restrict__ d, int nOut) {
  size_t i = (size_t)blockIdx.x * 256 + threadIdx.x;
  if (i >= (size_t)nOut * C_) return;
  int n = (int)(i >> 10), c = (int)(i & (C_ - 1));
  d[i] = f2bf(s[(size_t)c * nOut + n]);
}

// --------------------------------------------------------------- QKV GEMM --
// 128x128 block tile, 256 threads (8 waves: 4M x 2N, each wave 32x64), K=32,
// double-buffered LDS + one-deep async prefetch, one barrier per K-step.
__global__ __launch_bounds__(256) void qkv_gemm(const u16* __restrict__ xb,
                                                const u16* __restrict__ WaT,
                                                const float* __restrict__ bias,
                                                u16* __restrict__ qb,
                                                u16* __restrict__ kb,
                                                u16* __restrict__ vtb) {
  __shared__ u16 As[2][128][32];
  __shared__ u16 Bs[2][128][32];
  const int tid = threadIdx.x, lane = tid & 31, w = tid >> 5;
  const int wm = w >> 1, wn = w & 1;
  const int half = lane >> 4, l16 = lane & 15;
  const int bm = blockIdx.x * 128, bn = blockIdx.y * 128;
  const int lrow = tid >> 1, lc = (tid & 1) * 16;       // u16 col offset 0/16

  v8f acc[2][4] = {};
  const u16* arow = xb  + (size_t)(bm + lrow) * C_ + lc;
  const u16* brow = WaT + (size_t)(bn + lrow) * C_ + lc;
  const unsigned la[2] = { lds_off(&As[0][lrow][lc]), lds_off(&As[1][lrow][lc]) };
  const unsigned lb[2] = { lds_off(&Bs[0][lrow][lc]), lds_off(&Bs[1][lrow][lc]) };

  // prologue: prime buffer 0
  async_b128(la[0],      arow);
  async_b128(la[0] + 16, arow + 8);
  async_b128(lb[0],      brow);
  async_b128(lb[0] + 16, brow + 8);

  for (int i = 0; i < C_ / 32; ++i) {
    const int cur = i & 1, nxt = cur ^ 1;
    wait_async0();                    // buffer `cur` fully landed (this wave)
    __syncthreads();                  // ...for all waves; `nxt` reads done too
    if (i + 1 < C_ / 32) {            // prefetch next tile behind the WMMAs
      const u16* an = arow + (i + 1) * 32;
      const u16* bn2 = brow + (i + 1) * 32;
      async_b128(la[nxt],      an);
      async_b128(la[nxt] + 16, an + 8);
      async_b128(lb[nxt],      bn2);
      async_b128(lb[nxt] + 16, bn2 + 8);
    }

    Frag a[2], b[4];
#pragma unroll
    for (int mi = 0; mi < 2; ++mi) {
      const u16* p = &As[cur][wm * 32 + mi * 16 + l16][0];
      a[mi].q[0] = *(const uint4*)(p + 8 * half);
      a[mi].q[1] = *(const uint4*)(p + 16 + 8 * half);
    }
#pragma unroll
    for (int ni = 0; ni < 4; ++ni) {
      const u16* p = &Bs[cur][wn * 64 + ni * 16 + l16][0];
      b[ni].q[0] = *(const uint4*)(p + 8 * half);
      b[ni].q[1] = *(const uint4*)(p + 16 + 8 * half);
    }
#pragma unroll
    for (int mi = 0; mi < 2; ++mi)
#pragma unroll
      for (int ni = 0; ni < 4; ++ni)
        acc[mi][ni] = wmma_bf16(a[mi], b[ni], acc[mi][ni]);
  }

  // ---- epilogue: bias + scatter. Section / head / batch are UNIFORM per
  // wave (bn steps 128 inside 1024-wide sections; wave N-strip is 64-wide
  // inside one head; bm steps 128 inside T=2048), so hoist them. ----
  const int nbase = bn + wn * 64;              // 64-aligned
  const int sec   = nbase >> 10;               // 0=q 1=k 2=v (uniform)
  const int h     = (nbase & 1023) >> 6;       // head (uniform per wave)
  const int bbi   = bm >> 11;                  // batch (uniform)
  const int tb    = (bm & (T_ - 1)) + wm * 32; // t base for this wave
  const size_t bh = (size_t)(bbi * H_ + h);
  float bs[4];
#pragma unroll
  for (int ni = 0; ni < 4; ++ni) bs[ni] = bias[nbase + ni * 16 + l16];

  if (sec == 2) {                              // v^T[B,H,D,T]
    u16* dst = vtb + bh * D_ * T_;
#pragma unroll
    for (int mi = 0; mi < 2; ++mi)
#pragma unroll
      for (int ni = 0; ni < 4; ++ni)
#pragma unroll
        for (int r = 0; r < 8; ++r) {
          int t = tb + mi * 16 + r + 8 * half;
          dst[(size_t)(ni * 16 + l16) * T_ + t] = f2bf(acc[mi][ni][r] + bs[ni]);
        }
  } else {                                     // q/k[B,H,T,D]
    u16* dst = (sec == 0 ? qb : kb) + bh * T_ * D_;
#pragma unroll
    for (int mi = 0; mi < 2; ++mi)
#pragma unroll
      for (int ni = 0; ni < 4; ++ni)
#pragma unroll
        for (int r = 0; r < 8; ++r) {
          int t = tb + mi * 16 + r + 8 * half;
          dst[(size_t)t * D_ + ni * 16 + l16] = f2bf(acc[mi][ni][r] + bs[ni]);
        }
  }
}

// -------------------------------------------------------------- attention --
// One workgroup = 4 waves = one (b,h) x 64 query rows. Two passes over keys,
// double-buffered K/V tiles with one-deep async prefetch.
__global__ __launch_bounds__(128) void attn_kernel(const u16* __restrict__ qb,
                                                   const u16* __restrict__ kb,
                                                   const u16* __restrict__ vtb,
                                                   u16* __restrict__ ao) {
  __shared__ u16 Kt[2][64][64];   // [buf][key][d]
  __shared__ u16 Vt[2][64][64];   // [buf][d][key]
  __shared__ u16 Pb[4][16][64];   // per-wave P re-layout buffer
  const int tid = threadIdx.x, lane = tid & 31, w = tid >> 5;
  const int half = lane >> 4, l16 = lane & 15;
  const int bh = blockIdx.y;              // b*16 + h
  const int q0 = blockIdx.x * 64;
  const int lrow = tid >> 1, lh = tid & 1;
  const int NBLK = T_ / 64;

  // Q fragments for this wave's 16 rows (K = d, two 32-chunks) -- registers
  Frag qa[2];
  const u16* qrow = qb + ((size_t)bh * T_ + q0 + w * 16 + l16) * D_;
  qa[0].q[0] = *(const uint4*)(qrow + 8 * half);
  qa[0].q[1] = *(const uint4*)(qrow + 16 + 8 * half);
  qa[1].q[0] = *(const uint4*)(qrow + 32 + 8 * half);
  qa[1].q[1] = *(const uint4*)(qrow + 48 + 8 * half);

  const u16* kg = kb  + (size_t)bh * T_ * D_ + (size_t)lrow * D_ + lh * 32;
  const u16* vg = vtb + (size_t)bh * D_ * T_ + (size_t)lrow * T_ + lh * 32;
  const unsigned lk[2] = { lds_off(&Kt[0][lrow][lh * 32]),
                           lds_off(&Kt[1][lrow][lh * 32]) };
  const unsigned lv[2] = { lds_off(&Vt[0][lrow][lh * 32]),
                           lds_off(&Vt[1][lrow][lh * 32]) };

  // ---- pass 1: row sums of exp(S) ----
  float zsum[8] = {};
  async_b128(lk[0],      kg);
  async_b128(lk[0] + 16, kg + 8);
  async_b128(lk[0] + 32, kg + 16);
  async_b128(lk[0] + 48, kg + 24);
  for (int i = 0; i < NBLK; ++i) {
    const int cur = i & 1, nxt = cur ^ 1;
    wait_async0();
    __syncthreads();
    if (i + 1 < NBLK) {
      const u16* gn = kg + (size_t)(i + 1) * 64 * D_;
      async_b128(lk[nxt],      gn);
      async_b128(lk[nxt] + 16, gn + 8);
      async_b128(lk[nxt] + 32, gn + 16);
      async_b128(lk[nxt] + 48, gn + 24);
    }
#pragma unroll
    for (int ni = 0; ni < 4; ++ni) {
      Frag b0, b1;
      const u16* p = &Kt[cur][ni * 16 + l16][0];
      b0.q[0] = *(const uint4*)(p + 8 * half);
      b0.q[1] = *(const uint4*)(p + 16 + 8 * half);
      b1.q[0] = *(const uint4*)(p + 32 + 8 * half);
      b1.q[1] = *(const uint4*)(p + 48 + 8 * half);
      v8f s = {};
      s = wmma_bf16(qa[0], b0, s);
      s = wmma_bf16(qa[1], b1, s);
#pragma unroll
      for (int r = 0; r < 8; ++r) zsum[r] += __expf(s[r] * SCALE);
    }
  }
  // half-wave (16-lane) reduction: lanes 0-15 share rows 0-7, 16-31 rows 8-15
#pragma unroll
  for (int r = 0; r < 8; ++r) {
    float z = zsum[r];
    z += __shfl_xor(z, 1, 32);
    z += __shfl_xor(z, 2, 32);
    z += __shfl_xor(z, 4, 32);
    z += __shfl_xor(z, 8, 32);
    zsum[r] = __builtin_amdgcn_rcpf(z);   // Z^-1
  }
  __syncthreads();                        // pass-1 tile reads all done

  // ---- pass 2: P = relu(exp*Z^-1 - t); O += P @ V ----
  v8f oacc[4] = {};
  async_b128(lk[0],      kg);
  async_b128(lk[0] + 16, kg + 8);
  async_b128(lk[0] + 32, kg + 16);
  async_b128(lk[0] + 48, kg + 24);
  async_b128(lv[0],      vg);
  async_b128(lv[0] + 16, vg + 8);
  async_b128(lv[0] + 32, vg + 16);
  async_b128(lv[0] + 48, vg + 24);
  for (int i = 0; i < NBLK; ++i) {
    const int cur = i & 1, nxt = cur ^ 1;
    wait_async0();
    __syncthreads();
    if (i + 1 < NBLK) {
      const u16* gkn = kg + (size_t)(i + 1) * 64 * D_;
      const u16* gvn = vg + (i + 1) * 64;
      async_b128(lk[nxt],      gkn);
      async_b128(lk[nxt] + 16, gkn + 8);
      async_b128(lk[nxt] + 32, gkn + 16);
      async_b128(lk[nxt] + 48, gkn + 24);
      async_b128(lv[nxt],      gvn);
      async_b128(lv[nxt] + 16, gvn + 8);
      async_b128(lv[nxt] + 32, gvn + 16);
      async_b128(lv[nxt] + 48, gvn + 24);
    }

#pragma unroll
    for (int ni = 0; ni < 4; ++ni) {
      Frag b0, b1;
      const u16* p = &Kt[cur][ni * 16 + l16][0];
      b0.q[0] = *(const uint4*)(p + 8 * half);
      b0.q[1] = *(const uint4*)(p + 16 + 8 * half);
      b1.q[0] = *(const uint4*)(p + 32 + 8 * half);
      b1.q[1] = *(const uint4*)(p + 48 + 8 * half);
      v8f s = {};
      s = wmma_bf16(qa[0], b0, s);
      s = wmma_bf16(qa[1], b1, s);
#pragma unroll
      for (int r = 0; r < 8; ++r) {
        float pv = __expf(s[r] * SCALE) * zsum[r] - THRESH;
        pv = pv > 0.f ? pv : 0.f;
        Pb[w][r + 8 * half][ni * 16 + l16] = f2bf(pv);   // C/D -> row-major
      }
    }
    asm volatile("s_wait_dscnt 0" ::: "memory");         // P writes visible

    Frag pa[2];                                          // P as A fragments
    const u16* pp = &Pb[w][l16][0];
    pa[0].q[0] = *(const uint4*)(pp + 8 * half);
    pa[0].q[1] = *(const uint4*)(pp + 16 + 8 * half);
    pa[1].q[0] = *(const uint4*)(pp + 32 + 8 * half);
    pa[1].q[1] = *(const uint4*)(pp + 48 + 8 * half);

#pragma unroll
    for (int ni = 0; ni < 4; ++ni) {
      Frag v0, v1;
      const u16* vp = &Vt[cur][ni * 16 + l16][0];
      v0.q[0] = *(const uint4*)(vp + 8 * half);
      v0.q[1] = *(const uint4*)(vp + 16 + 8 * half);
      v1.q[0] = *(const uint4*)(vp + 32 + 8 * half);
      v1.q[1] = *(const uint4*)(vp + 48 + 8 * half);
      oacc[ni] = wmma_bf16(pa[0], v0, oacc[ni]);
      oacc[ni] = wmma_bf16(pa[1], v1, oacc[ni]);
    }
  }

  // write O (bf16) back into [B,T,C] layout for the projection GEMM
  const int bb = bh >> 4, h = bh & 15;
  u16* dst = ao + (size_t)bb * T_ * C_ + (size_t)h * D_;
#pragma unroll
  for (int ni = 0; ni < 4; ++ni)
#pragma unroll
    for (int r = 0; r < 8; ++r) {
      int t = q0 + w * 16 + r + 8 * half;
      dst[(size_t)t * C_ + ni * 16 + l16] = f2bf(oacc[ni][r]);
    }
}

// -------------------------------------------------------------- proj GEMM --
__global__ __launch_bounds__(256) void proj_gemm(const u16* __restrict__ ain,
                                                 const u16* __restrict__ WpT,
                                                 const float* __restrict__ bias,
                                                 float* __restrict__ out) {
  __shared__ u16 As[2][128][32];
  __shared__ u16 Bs[2][128][32];
  const int tid = threadIdx.x, lane = tid & 31, w = tid >> 5;
  const int wm = w >> 1, wn = w & 1;
  const int half = lane >> 4, l16 = lane & 15;
  const int bm = blockIdx.x * 128, bn = blockIdx.y * 128;
  const int lrow = tid >> 1, lc = (tid & 1) * 16;

  v8f acc[2][4] = {};
  const u16* arow = ain + (size_t)(bm + lrow) * C_ + lc;
  const u16* brow = WpT + (size_t)(bn + lrow) * C_ + lc;
  const unsigned la[2] = { lds_off(&As[0][lrow][lc]), lds_off(&As[1][lrow][lc]) };
  const unsigned lb[2] = { lds_off(&Bs[0][lrow][lc]), lds_off(&Bs[1][lrow][lc]) };

  async_b128(la[0],      arow);
  async_b128(la[0] + 16, arow + 8);
  async_b128(lb[0],      brow);
  async_b128(lb[0] + 16, brow + 8);

  for (int i = 0; i < C_ / 32; ++i) {
    const int cur = i & 1, nxt = cur ^ 1;
    wait_async0();
    __syncthreads();
    if (i + 1 < C_ / 32) {
      const u16* an = arow + (i + 1) * 32;
      const u16* bn2 = brow + (i + 1) * 32;
      async_b128(la[nxt],      an);
      async_b128(la[nxt] + 16, an + 8);
      async_b128(lb[nxt],      bn2);
      async_b128(lb[nxt] + 16, bn2 + 8);
    }

    Frag a[2], b[4];
#pragma unroll
    for (int mi = 0; mi < 2; ++mi) {
      const u16* p = &As[cur][wm * 32 + mi * 16 + l16][0];
      a[mi].q[0] = *(const uint4*)(p + 8 * half);
      a[mi].q[1] = *(const uint4*)(p + 16 + 8 * half);
    }
#pragma unroll
    for (int ni = 0; ni < 4; ++ni) {
      const u16* p = &Bs[cur][wn * 64 + ni * 16 + l16][0];
      b[ni].q[0] = *(const uint4*)(p + 8 * half);
      b[ni].q[1] = *(const uint4*)(p + 16 + 8 * half);
    }
#pragma unroll
    for (int mi = 0; mi < 2; ++mi)
#pragma unroll
      for (int ni = 0; ni < 4; ++ni)
        acc[mi][ni] = wmma_bf16(a[mi], b[ni], acc[mi][ni]);
  }

  const int tb = bm + wm * 32;
  float bs[4];
#pragma unroll
  for (int ni = 0; ni < 4; ++ni) bs[ni] = bias[bn + wn * 64 + ni * 16 + l16];
#pragma unroll
  for (int mi = 0; mi < 2; ++mi)
#pragma unroll
    for (int ni = 0; ni < 4; ++ni)
#pragma unroll
      for (int r = 0; r < 8; ++r) {
        int m = tb + mi * 16 + r + 8 * half;
        int n = bn + wn * 64 + ni * 16 + l16;
        out[(size_t)m * C_ + n] = acc[mi][ni][r] + bs[ni];
      }
}

// ------------------------------------------------------------------ launch --
extern "C" void kernel_launch(void* const* d_in, const int* in_sizes, int n_in,
                              void* d_out, int out_size, void* d_ws, size_t ws_size,
                              hipStream_t stream) {
  const float* x      = (const float*)d_in[0];
  const float* W_attn = (const float*)d_in[1];
  const float* b_attn = (const float*)d_in[2];
  const float* W_proj = (const float*)d_in[3];
  const float* b_proj = (const float*)d_in[4];
  float* out = (float*)d_out;

  u16* ws  = (u16*)d_ws;
  u16* xb  = ws + OFF_XB;
  u16* WaT = ws + OFF_WAT;
  u16* WpT = ws + OFF_WPT;
  u16* qb  = ws + OFF_Q;
  u16* kb  = ws + OFF_K;
  u16* vtb = ws + OFF_VT;
  u16* ao  = ws + OFF_AO;

  cvt_f32_bf16<<<dim3((unsigned)(N_X / 256)), 256, 0, stream>>>(x, xb, N_X);
  cvt_transpose<<<dim3((unsigned)(N_WA / 256)), 256, 0, stream>>>(W_attn, WaT, 3 * C_);
  cvt_transpose<<<dim3((unsigned)(N_WP / 256)), 256, 0, stream>>>(W_proj, WpT, C_);

  qkv_gemm<<<dim3(B_ * T_ / 128, 3 * C_ / 128), 256, 0, stream>>>(xb, WaT, b_attn,
                                                                  qb, kb, vtb);
  attn_kernel<<<dim3(T_ / 64, B_ * H_), 128, 0, stream>>>(qb, kb, vtb, ao);
  proj_gemm<<<dim3(B_ * T_ / 128, C_ / 128), 256, 0, stream>>>(ao, WpT, b_proj, out);
}
